// LatentCrossAttentionBlock_9457517986332
// MI455X (gfx1250) — compile-verified, matrix operand order checked by reading, not run on hardware
//
#include <hip/hip_runtime.h>
#include <hip/hip_bf16.h>
#include <math.h>

// Problem dims (LatentCrossAttentionBlock)
#define BDIM 8
#define LDIM 64
#define SDIM 2048
#define DDIM 1024
#define CDIM 1024
#define HDIM 16
#define HDHD 64      // head dim = D/H
#define FFDIM 4096   // 4*D

typedef __attribute__((ext_vector_type(16))) __bf16 v16bf;
typedef __attribute__((ext_vector_type(8)))  __bf16 v8bf;
typedef __attribute__((ext_vector_type(8)))  float  v8f;

// fp32 -> bf16, round-to-nearest-even
__device__ __forceinline__ __bf16 f2bf(float f) {
    union { float f; unsigned u; } in; in.f = f;
    unsigned u = in.u;
    u += 0x7FFFu + ((u >> 16) & 1u);
    union { unsigned short s; __bf16 b; } out;
    out.s = (unsigned short)(u >> 16);
    return out.b;
}

// bf16 row fragment: elements 0..7 = p[0..7], 8..15 = p[16..23]
// -> two 16-byte loads, zero conversion work in the GEMM loop
__device__ __forceinline__ v16bf load_frag_row_bf(const __bf16* __restrict__ p) {
    v8bf lo = *(const v8bf*)(p);
    v8bf hi = *(const v8bf*)(p + 16);
    return __builtin_shufflevector(lo, hi, 0,1,2,3,4,5,6,7,8,9,10,11,12,13,14,15);
}

// Batched GEMM on bf16 operands, f32 accumulate:
//   C[m,n] = alpha * sum_k A[m,k] * W[n,k] + bias[n]   (A row-major, W[N,K] row-major)
//   MT: M-tiles (16 rows each) per wave -- B fragments reused across MT WMMAs
//   EPI: 0 = none, 1 = exact GELU
//   OUTMODE: 0 = f32, 1 = bf16, 2 = bf16 transposed-per-head ([B*S,D] -> [B,H,HD,S])
// Block = 128 threads = 4 waves; each wave computes a (16*MT)x64 strip via
// 4*MT v_wmma_f32_16x16x32_bf16 per 32-wide K step. (b,h) batch via blockIdx.z.
template <int MT, int EPI, int OUTMODE>
__global__ __launch_bounds__(128)
void gemm_wmma_bf16(const __bf16* __restrict__ A, int lda, long long sAb, long long sAh,
                    const __bf16* __restrict__ Bm, int ldb, long long sBb, long long sBh,
                    const float* __restrict__ bias,
                    void* __restrict__ Cv, int ldc, long long sCb, long long sCh,
                    int M, int N, int K, float alpha, int hdim)
{
    const int lane = threadIdx.x & 31;
    const int wave = threadIdx.x >> 5;
    const int half = lane >> 4;     // lane-group (ISA K split)
    const int l16  = lane & 15;

    const int z  = blockIdx.z;
    const int bz = z / hdim;
    const int hz = z - bz * hdim;
    A  += bz * sAb + hz * sAh;
    Bm += bz * sBb + hz * sBh;
    const long long cbase = bz * sCb + hz * sCh;

    const int m0 = blockIdx.y * (64 * MT) + wave * (16 * MT);
    const int n0 = blockIdx.x * 64;
    if (m0 >= M) return;

    v8f acc[MT][4] = {};

    const __bf16* arow[MT];
#pragma unroll
    for (int i = 0; i < MT; ++i)
        arow[i] = A + (size_t)(m0 + i * 16 + l16) * lda;
    const __bf16* brow = Bm + (size_t)(n0 + l16) * ldb;

    for (int k0 = 0; k0 < K; k0 += 32) {
        v16bf afrag[MT];
#pragma unroll
        for (int i = 0; i < MT; ++i)
            afrag[i] = load_frag_row_bf(arow[i] + k0 + 8 * half);
#pragma unroll
        for (int t = 0; t < 4; ++t) {
            const v16bf bfrag = load_frag_row_bf(brow + (size_t)t * 16 * ldb + k0 + 8 * half);
#pragma unroll
            for (int i = 0; i < MT; ++i)
                acc[i][t] = __builtin_amdgcn_wmma_f32_16x16x32_bf16(
                    false, afrag[i], false, bfrag, (short)0, acc[i][t], false, false);
        }
    }

    // C/D layout: element r -> row (m-tile base) + r + 8*half, col n0 + 16*t + l16
#pragma unroll
    for (int t = 0; t < 4; ++t) {
        const int n = n0 + t * 16 + l16;
        const float bv = bias ? bias[n] : 0.0f;
#pragma unroll
        for (int i = 0; i < MT; ++i) {
#pragma unroll
            for (int r = 0; r < 8; ++r) {
                const int m = m0 + i * 16 + r + 8 * half;
                float v = acc[i][t][r] * alpha + bv;
                if (EPI == 1) v = 0.5f * v * (1.0f + erff(v * 0.70710678118654752f));
                if (OUTMODE == 0) {
                    ((float*)Cv)[(size_t)cbase + (size_t)m * ldc + n] = v;
                } else if (OUTMODE == 1) {
                    ((__bf16*)Cv)[(size_t)cbase + (size_t)m * ldc + n] = f2bf(v);
                } else {
                    // [B*S, D] value (m,n) -> vhT[b][h][dd][s], s contiguous
                    const int b  = m / SDIM;
                    const int s  = m - b * SDIM;
                    const int h  = n >> 6;      // n / HDHD
                    const int dd = n & 63;      // n % HDHD
                    ((__bf16*)Cv)[(((size_t)b * HDIM + h) * HDHD + dd) * SDIM + s] = f2bf(v);
                }
            }
        }
    }
}

// One-shot fp32 -> bf16 matrix mirror (8 elements / thread, b128 in, b128 out)
__global__ __launch_bounds__(256)
void cvt_f32_bf16_kernel(const float* __restrict__ in, __bf16* __restrict__ out,
                         long long n)
{
    const long long i = ((long long)blockIdx.x * 256 + threadIdx.x) * 8;
    if (i >= n) return;
    const float4* p = (const float4*)(in + i);
    const float4 a = p[0], b = p[1];
    v8bf r;
    r[0] = f2bf(a.x); r[1] = f2bf(a.y); r[2] = f2bf(a.z); r[3] = f2bf(a.w);
    r[4] = f2bf(b.x); r[5] = f2bf(b.y); r[6] = f2bf(b.z); r[7] = f2bf(b.w);
    *(v8bf*)(out + i) = r;
}

// Tiled transpose + convert: out[c*rows + r] = bf16(in[r*cols + c])
__global__ __launch_bounds__(256)
void transpose_cvt_kernel(const float* __restrict__ in, __bf16* __restrict__ out,
                          int rows, int cols)
{
    __shared__ float tile[32][33];
    const int tx = threadIdx.x & 31;
    const int ty = threadIdx.x >> 5;            // 0..7
    const int r0 = blockIdx.y * 32;
    const int c0 = blockIdx.x * 32;
#pragma unroll
    for (int i = 0; i < 32; i += 8)
        tile[ty + i][tx] = in[(size_t)(r0 + ty + i) * cols + (c0 + tx)];
    __syncthreads();
#pragma unroll
    for (int i = 0; i < 32; i += 8)
        out[(size_t)(c0 + ty + i) * rows + (r0 + tx)] = f2bf(tile[tx][ty + i]);
}

// Combined bias: out[o] = sum_j in_w[o,j]*base_b[j] + in_b[o]
__global__ __launch_bounds__(256)
void bias_combine_kernel(const float* __restrict__ inw, const float* __restrict__ baseb,
                         const float* __restrict__ inb, float* __restrict__ out)
{
    const int o = blockIdx.x * 256 + threadIdx.x;
    if (o >= DDIM) return;
    const float* wr = inw + (size_t)o * DDIM;
    float acc = inb[o];
    for (int j = 0; j < DDIM; ++j) acc += wr[j] * baseb[j];
    out[o] = acc;
}

// In-place masked softmax over S for one (b,h,l) row per block,
// plus bf16 mirror of the weights for the attn GEMM.
__global__ __launch_bounds__(256)
void softmax_mask_kernel(float* __restrict__ scores, __bf16* __restrict__ wbf,
                         const int* __restrict__ mask)
{
    __shared__ float red[256];
    const int bid = blockIdx.x;                 // (b*H + h)*L + l
    const int b   = bid / (HDIM * LDIM);
    float* row = scores + (size_t)bid * SDIM;
    __bf16* brow = wbf + (size_t)bid * SDIM;
    const int* mrow = mask + (size_t)b * SDIM;
    const int tid = threadIdx.x;

    float lmax = -3.4e38f;
    for (int s = tid; s < SDIM; s += 256)
        if (mrow[s] != 0) lmax = fmaxf(lmax, row[s]);
    red[tid] = lmax; __syncthreads();
    for (int w = 128; w > 0; w >>= 1) {
        if (tid < w) red[tid] = fmaxf(red[tid], red[tid + w]);
        __syncthreads();
    }
    const float gmax = red[0]; __syncthreads();

    float lsum = 0.0f;
    for (int s = tid; s < SDIM; s += 256) {
        const float e = (mrow[s] != 0) ? __expf(row[s] - gmax) : 0.0f;
        row[s] = e; lsum += e;
    }
    red[tid] = lsum; __syncthreads();
    for (int w = 128; w > 0; w >>= 1) {
        if (tid < w) red[tid] += red[tid + w];
        __syncthreads();
    }
    const float inv = red[0] > 0.0f ? 1.0f / red[0] : 0.0f;
    for (int s = tid; s < SDIM; s += 256) {
        const float v = row[s] * inv;
        row[s] = v;
        brow[s] = f2bf(v);
    }
}

// attn_weights[b,l,s] = mean_h w[b,h,l,s]
__global__ __launch_bounds__(256)
void head_mean_kernel(const float* __restrict__ w, float* __restrict__ out)
{
    const size_t idx = (size_t)blockIdx.x * 256 + threadIdx.x;
    const size_t total = (size_t)BDIM * LDIM * SDIM;
    if (idx >= total) return;
    const size_t b   = idx / ((size_t)LDIM * SDIM);
    const size_t rem = idx - b * (size_t)LDIM * SDIM;
    float s = 0.0f;
    for (int h = 0; h < HDIM; ++h)
        s += w[(b * HDIM + h) * (size_t)LDIM * SDIM + rem];
    out[idx] = s * (1.0f / HDIM);
}

// out[r,:] = LayerNorm(a[r,:] + b[r,:]) * g + be  (+ optional bf16 mirror)
__global__ __launch_bounds__(256)
void add_layernorm_kernel(const float* __restrict__ a, const float* __restrict__ b,
                          const float* __restrict__ g, const float* __restrict__ be,
                          float* __restrict__ out, __bf16* __restrict__ outbf)
{
    __shared__ float red[256], red2[256];
    const int r = blockIdx.x;
    const float* ar = a + (size_t)r * DDIM;
    const float* br = b + (size_t)r * DDIM;
    float* orow = out + (size_t)r * DDIM;
    const int tid = threadIdx.x;

    float s = 0.0f, s2 = 0.0f;
    for (int i = tid; i < DDIM; i += 256) {
        const float x = ar[i] + br[i];
        s += x; s2 += x * x;
    }
    red[tid] = s; red2[tid] = s2; __syncthreads();
    for (int w = 128; w > 0; w >>= 1) {
        if (tid < w) { red[tid] += red[tid + w]; red2[tid] += red2[tid + w]; }
        __syncthreads();
    }
    const float mean = red[0] * (1.0f / DDIM);
    const float var  = red2[0] * (1.0f / DDIM) - mean * mean;
    const float rstd = rsqrtf(var + 1e-5f);
    for (int i = tid; i < DDIM; i += 256) {
        const float x = ar[i] + br[i];
        const float y = (x - mean) * rstd * g[i] + be[i];
        orow[i] = y;
        if (outbf) outbf[(size_t)r * DDIM + i] = f2bf(y);
    }
}

static inline dim3 cvt_grid(long long n) { return dim3((unsigned)((n / 8 + 255) / 256)); }

extern "C" void kernel_launch(void* const* d_in, const int* in_sizes, int n_in,
                              void* d_out, int out_size, void* d_ws, size_t ws_size,
                              hipStream_t stream)
{
    (void)in_sizes; (void)n_in; (void)out_size; (void)ws_size;

    const float* latents = (const float*)d_in[0];
    const float* context = (const float*)d_in[1];
    const int*   cmask   = (const int*)d_in[2];
    const float* q_w   = (const float*)d_in[3];  const float* q_b   = (const float*)d_in[4];
    const float* k_w   = (const float*)d_in[5];  const float* k_b   = (const float*)d_in[6];
    const float* v_w   = (const float*)d_in[7];  const float* v_b   = (const float*)d_in[8];
    const float* in_wq = (const float*)d_in[9];  const float* in_bq = (const float*)d_in[10];
    const float* in_wk = (const float*)d_in[11]; const float* in_bk = (const float*)d_in[12];
    const float* in_wv = (const float*)d_in[13]; const float* in_bv = (const float*)d_in[14];
    const float* out_w = (const float*)d_in[15]; const float* out_b = (const float*)d_in[16];
    const float* ln1_g = (const float*)d_in[17]; const float* ln1_b = (const float*)d_in[18];
    const float* ln2_g = (const float*)d_in[19]; const float* ln2_b = (const float*)d_in[20];
    const float* ff_w1 = (const float*)d_in[21]; const float* ff_b1 = (const float*)d_in[22];
    const float* ff_w2 = (const float*)d_in[23]; const float* ff_b2 = (const float*)d_in[24];

    // Byte-based workspace allocator, 16B-aligned regions
    char* base = (char*)d_ws;
    size_t off = 0;
    auto allocf = [&](size_t n) { float* p = (float*)(base + off);
                                  off = (off + n * 4 + 15) & ~(size_t)15; return p; };
    auto allocb = [&](size_t n) { __bf16* p = (__bf16*)(base + off);
                                  off = (off + n * 2 + 15) & ~(size_t)15; return p; };

    const size_t nLat = (size_t)BDIM * LDIM * DDIM;        // 512K
    const size_t nCtx = (size_t)BDIM * SDIM * CDIM;        // 16M
    const size_t nW   = (size_t)DDIM * DDIM;               // 1M
    const size_t nKV  = (size_t)BDIM * SDIM * DDIM;        // 16M
    const size_t nSc  = (size_t)BDIM * HDIM * LDIM * SDIM; // 16M

    // f32 buffers
    float* bqc = allocf(DDIM);
    float* bkc = allocf(DDIM);
    float* bvc = allocf(DDIM);
    float* sc  = allocf(nSc);          // scores / softmax weights (f32)
    float* tmp = allocf(nLat);         // attn_out
    float* xln = allocf(nLat);         // x after LN1
    float* ffo = allocf(nLat);         // FF output

    // bf16 mirrors of inputs (base projection weights transposed for row access)
    __bf16* lat_bf  = allocb(nLat);
    __bf16* ctx_bf  = allocb(nCtx);
    __bf16* iwq_bf  = allocb(nW);
    __bf16* iwk_bf  = allocb(nW);
    __bf16* iwv_bf  = allocb(nW);
    __bf16* qwT_bf  = allocb(nW);   // q_w^T  [C,D]
    __bf16* kwT_bf  = allocb(nW);   // k_w^T
    __bf16* vwT_bf  = allocb(nW);   // v_w^T
    __bf16* outw_bf = allocb(nW);
    __bf16* fw1_bf  = allocb((size_t)FFDIM * DDIM);
    __bf16* fw2_bf  = allocb((size_t)DDIM * FFDIM);
    // bf16 intermediates
    __bf16* wqc_bf  = allocb(nW);
    __bf16* wkc_bf  = allocb(nW);
    __bf16* wvc_bf  = allocb(nW);
    __bf16* qh_bf   = allocb(nLat);
    __bf16* kh_bf   = allocb(nKV);
    __bf16* vhT_bf  = allocb(nKV);  // V transposed per head: [B,H,HD,S]
    __bf16* w_bf    = allocb(nSc);
    __bf16* attn_bf = allocb(nLat);
    __bf16* xln_bf  = allocb(nLat);
    __bf16* hid_bf  = allocb((size_t)BDIM * LDIM * FFDIM);

    float* outp = (float*)d_out;                      // [B,L,D]
    float* attw = outp + nLat;                        // [B,L,S]

    const dim3 blk(128);

    // 0) one-shot bf16 mirrors / transposed mirrors of every matrix operand
    cvt_f32_bf16_kernel<<<cvt_grid(nLat), 256, 0, stream>>>(latents, lat_bf, nLat);
    cvt_f32_bf16_kernel<<<cvt_grid(nCtx), 256, 0, stream>>>(context, ctx_bf, nCtx);
    cvt_f32_bf16_kernel<<<cvt_grid(nW), 256, 0, stream>>>(in_wq, iwq_bf, nW);
    cvt_f32_bf16_kernel<<<cvt_grid(nW), 256, 0, stream>>>(in_wk, iwk_bf, nW);
    cvt_f32_bf16_kernel<<<cvt_grid(nW), 256, 0, stream>>>(in_wv, iwv_bf, nW);
    transpose_cvt_kernel<<<dim3(CDIM / 32, DDIM / 32), 256, 0, stream>>>(q_w, qwT_bf, DDIM, CDIM);
    transpose_cvt_kernel<<<dim3(CDIM / 32, DDIM / 32), 256, 0, stream>>>(k_w, kwT_bf, DDIM, CDIM);
    transpose_cvt_kernel<<<dim3(CDIM / 32, DDIM / 32), 256, 0, stream>>>(v_w, vwT_bf, DDIM, CDIM);
    cvt_f32_bf16_kernel<<<cvt_grid(nW), 256, 0, stream>>>(out_w, outw_bf, nW);
    cvt_f32_bf16_kernel<<<cvt_grid((size_t)FFDIM * DDIM), 256, 0, stream>>>(ff_w1, fw1_bf, (long long)FFDIM * DDIM);
    cvt_f32_bf16_kernel<<<cvt_grid((size_t)DDIM * FFDIM), 256, 0, stream>>>(ff_w2, fw2_bf, (long long)DDIM * FFDIM);

    // 1) Combined biases: bc = in_w @ base_b + in_b
    bias_combine_kernel<<<dim3(DDIM / 256), 256, 0, stream>>>(in_wq, q_b, in_bq, bqc);
    bias_combine_kernel<<<dim3(DDIM / 256), 256, 0, stream>>>(in_wk, k_b, in_bk, bkc);
    bias_combine_kernel<<<dim3(DDIM / 256), 256, 0, stream>>>(in_wv, v_b, in_bv, bvc);

    // 2) Combined weights: Wc[o,i] = sum_j in_w[o,j] * baseT[i,j]  -> bf16
    gemm_wmma_bf16<2, 0, 1><<<dim3(CDIM / 64, DDIM / 128, 1), blk, 0, stream>>>(
        iwq_bf, DDIM, 0, 0, qwT_bf, DDIM, 0, 0, nullptr, wqc_bf, CDIM, 0, 0,
        DDIM, CDIM, DDIM, 1.0f, 1);
    gemm_wmma_bf16<2, 0, 1><<<dim3(CDIM / 64, DDIM / 128, 1), blk, 0, stream>>>(
        iwk_bf, DDIM, 0, 0, kwT_bf, DDIM, 0, 0, nullptr, wkc_bf, CDIM, 0, 0,
        DDIM, CDIM, DDIM, 1.0f, 1);
    gemm_wmma_bf16<2, 0, 1><<<dim3(CDIM / 64, DDIM / 128, 1), blk, 0, stream>>>(
        iwv_bf, DDIM, 0, 0, vwT_bf, DDIM, 0, 0, nullptr, wvc_bf, CDIM, 0, 0,
        DDIM, CDIM, DDIM, 1.0f, 1);

    // 3) qh = latents @ Wqc^T + bqc  -> bf16
    gemm_wmma_bf16<2, 0, 1><<<dim3(DDIM / 64, (BDIM * LDIM) / 128, 1), blk, 0, stream>>>(
        lat_bf, DDIM, 0, 0, wqc_bf, DDIM, 0, 0, bqc, qh_bf, DDIM, 0, 0,
        BDIM * LDIM, DDIM, DDIM, 1.0f, 1);

    // 4) kh = context @ Wkc^T + bkc -> bf16 ; vh -> bf16 transposed per head
    gemm_wmma_bf16<2, 0, 1><<<dim3(DDIM / 64, (BDIM * SDIM) / 128, 1), blk, 0, stream>>>(
        ctx_bf, CDIM, 0, 0, wkc_bf, CDIM, 0, 0, bkc, kh_bf, DDIM, 0, 0,
        BDIM * SDIM, DDIM, CDIM, 1.0f, 1);
    gemm_wmma_bf16<2, 0, 2><<<dim3(DDIM / 64, (BDIM * SDIM) / 128, 1), blk, 0, stream>>>(
        ctx_bf, CDIM, 0, 0, wvc_bf, CDIM, 0, 0, bvc, vhT_bf, DDIM, 0, 0,
        BDIM * SDIM, DDIM, CDIM, 1.0f, 1);

    // 5) scores[b,h] = (qh @ kh^T)/8  -> f32, batched over B*H  (M=64 -> MT=1)
    gemm_wmma_bf16<1, 0, 0><<<dim3(SDIM / 64, 1, BDIM * HDIM), blk, 0, stream>>>(
        qh_bf, DDIM, (long long)LDIM * DDIM, HDHD,
        kh_bf, DDIM, (long long)SDIM * DDIM, HDHD,
        nullptr,
        sc, SDIM, (long long)HDIM * LDIM * SDIM, (long long)LDIM * SDIM,
        LDIM, SDIM, HDHD, 0.125f, HDIM);

    // 6) masked softmax in place (f32) + bf16 mirror for the attn GEMM
    softmax_mask_kernel<<<dim3(BDIM * HDIM * LDIM), 256, 0, stream>>>(sc, w_bf, cmask);

    // 7) attn_weights = mean over heads -> second output region
    head_mean_kernel<<<dim3((BDIM * LDIM * SDIM) / 256), 256, 0, stream>>>(sc, attw);

    // 8) attn[b,l,h,:] = w[b,h] @ vhT[b,h]^T  (row-major W[N=HD,K=S]), batched -> bf16
    gemm_wmma_bf16<1, 0, 1><<<dim3(1, 1, BDIM * HDIM), blk, 0, stream>>>(
        w_bf, SDIM, (long long)HDIM * LDIM * SDIM, (long long)LDIM * SDIM,
        vhT_bf, SDIM, (long long)HDIM * HDHD * SDIM, (long long)HDHD * SDIM,
        nullptr,
        attn_bf, DDIM, (long long)LDIM * DDIM, HDHD,
        LDIM, HDHD, SDIM, 1.0f, HDIM);

    // 9) attn_out = attn @ out_w^T + out_b -> f32
    gemm_wmma_bf16<2, 0, 0><<<dim3(DDIM / 64, (BDIM * LDIM) / 128, 1), blk, 0, stream>>>(
        attn_bf, DDIM, 0, 0, outw_bf, DDIM, 0, 0, out_b, tmp, DDIM, 0, 0,
        BDIM * LDIM, DDIM, DDIM, 1.0f, 1);

    // 10) x = LN1(latents + attn_out) -> f32 + bf16
    add_layernorm_kernel<<<dim3(BDIM * LDIM), 256, 0, stream>>>(latents, tmp, ln1_g, ln1_b, xln, xln_bf);

    // 11) hidden = GELU(x @ ff_w1^T + ff_b1) -> bf16
    gemm_wmma_bf16<2, 1, 1><<<dim3(FFDIM / 64, (BDIM * LDIM) / 128, 1), blk, 0, stream>>>(
        xln_bf, DDIM, 0, 0, fw1_bf, DDIM, 0, 0, ff_b1, hid_bf, FFDIM, 0, 0,
        BDIM * LDIM, FFDIM, DDIM, 1.0f, 1);

    // 12) ff = hidden @ ff_w2^T + ff_b2 -> f32
    gemm_wmma_bf16<2, 0, 0><<<dim3(DDIM / 64, (BDIM * LDIM) / 128, 1), blk, 0, stream>>>(
        hid_bf, FFDIM, 0, 0, fw2_bf, FFDIM, 0, 0, ff_b2, ffo, DDIM, 0, 0,
        BDIM * LDIM, DDIM, FFDIM, 1.0f, 1);

    // 13) out = LN2(x + ff) -> first output region
    add_layernorm_kernel<<<dim3(BDIM * LDIM), 256, 0, stream>>>(xln, ffo, ln2_g, ln2_b, outp, nullptr);
}
